// DKVMN_CORE_48515950576234
// MI455X (gfx1250) — compile-verified, matrix operand order checked by reading, not run on hardware
//
#include <hip/hip_runtime.h>
#include <hip/hip_bf16.h>
#include <math.h>

// ---------- WMMA types ----------
typedef __attribute__((ext_vector_type(16))) _Float16 v16h;
typedef __attribute__((ext_vector_type(8)))  float    v8f;

#define EPSV 1e-12f

// A in LDS, row-major f16, row stride astride. B in global, pre-transposed:
// Bt[n][k], row stride bstride (== Kdim). One wave computes one 16x16 tile.
__device__ __forceinline__ void wmma_tile(const _Float16* __restrict__ Alds, int astride,
                                          const _Float16* __restrict__ Bt, int bstride,
                                          int Kdim, int ntile, int lane, v8f& acc) {
    const int m  = lane & 15;        // A row / B col within tile
    const int hi = (lane >> 4) * 16; // K sub-block select (contiguous k = 16h + i)
    for (int kc = 0; kc < Kdim; kc += 32) {
        v16h a, b;
        const _Float16* ap = Alds + m * astride + kc + hi;
        const _Float16* bp = Bt + (size_t)(ntile * 16 + m) * bstride + kc + hi;
#pragma unroll
        for (int i = 0; i < 16; ++i) { a[i] = ap[i]; b[i] = bp[i]; }
        acc = __builtin_amdgcn_wmma_f32_16x16x32_f16(false, a, false, b,
                                                     (short)0, acc, false, false);
    }
}

__device__ __forceinline__ float sigm(float x) { return 1.f / (1.f + expf(-x)); }

__device__ __forceinline__ void prefetch_bytes(const void* p, int nbytes, int tid, int nthr) {
    const char* c = (const char*)p;
    for (int off = tid * 64; off < nbytes; off += nthr * 64)
        __builtin_prefetch(c + off, 0, 0);   // -> global_prefetch_b8
}

// ---------- 1. weight prep: f32 -> f16, transposed to Bt[n][k] ----------
__global__ void dkvmn_prep(const float* __restrict__ Wk, const float* __restrict__ Wv,
                           const float* __restrict__ We, const float* __restrict__ Wa,
                           const float* __restrict__ Wf, const float* __restrict__ Mk,
                           _Float16* WkT, _Float16* WvT, _Float16* WeT,
                           _Float16* WaT, _Float16* WfT, _Float16* MkT) {
    int tid = blockIdx.x * blockDim.x + threadIdx.x;
    int stride = gridDim.x * blockDim.x;
    for (int i = tid; i < 64 * 128; i += stride) { int n = i >> 7, k = i & 127; WkT[i] = (_Float16)Wk[k * 64 + n]; }
    for (int i = tid; i < 64 * 256; i += stride) { int n = i >> 8, k = i & 255; WvT[i] = (_Float16)Wv[k * 64 + n]; }
    for (int i = tid; i < 64 * 64;  i += stride) {
        int n = i >> 6, k = i & 63;
        WeT[i] = (_Float16)We[k * 64 + n];
        WaT[i] = (_Float16)Wa[k * 64 + n];
        MkT[i] = (n < 50) ? (_Float16)Mk[n * 64 + k] : (_Float16)0.f;  // Bt[n][k] = Mk[n][k], pad N 50->64
    }
    for (int i = tid; i < 64 * 128; i += stride) { int n = i >> 7, k = i & 127; WfT[i] = (_Float16)Wf[k * 64 + n]; }
}

// ---------- 2. embed + front GEMMs (k, v, e, a, w) ----------
// 2048 blocks x 128 threads; 16 rows per block, 4 waves = 4 N-tiles of 16 cols.
__global__ void dkvmn_front(const int* __restrict__ qseqs, const int* __restrict__ cseqs,
                            const int* __restrict__ rseqs, const int* __restrict__ sqseqs,
                            const int* __restrict__ scseqs, const int* __restrict__ srseqs,
                            const float* __restrict__ qtab, const float* __restrict__ ctab,
                            const _Float16* __restrict__ WkT, const _Float16* __restrict__ WvT,
                            const _Float16* __restrict__ WeT, const _Float16* __restrict__ WaT,
                            const _Float16* __restrict__ MkT,
                            const float* __restrict__ bk, const float* __restrict__ bv,
                            const float* __restrict__ be, const float* __restrict__ ba,
                            float* __restrict__ kf, _Float16* __restrict__ khg,
                            float* __restrict__ ef, float* __restrict__ af,
                            float* __restrict__ wf) {
    __shared__ _Float16 xh[16 * 128];
    __shared__ _Float16 eh[16 * 256];
    __shared__ _Float16 khl[16 * 64];
    __shared__ _Float16 vhl[16 * 64];
    __shared__ float sc[16 * 64];
    __shared__ float rr_s[16];

    const int tid = threadIdx.x, lane = tid & 31, wid = tid >> 5;
    const int row0 = blockIdx.x * 16;

    // warm the weight images while we gather embeddings
    prefetch_bytes(WkT, 64 * 128 * 2, tid, blockDim.x);
    prefetch_bytes(WvT, 64 * 256 * 2, tid, blockDim.x);
    prefetch_bytes(WeT, 64 * 64 * 2, tid, blockDim.x);
    prefetch_bytes(WaT, 64 * 64 * 2, tid, blockDim.x);
    prefetch_bytes(MkT, 64 * 64 * 2, tid, blockDim.x);

    if (tid < 16) {
        int row = row0 + tid, b = row >> 9, t = row & 511;
        rr_s[tid] = (float)((t == 0) ? rseqs[b * 511] : srseqs[b * 511 + t - 1]);
    }
    // x = [q_e | mean(c_e)] as f16
    for (int idx = tid; idx < 16 * 128; idx += blockDim.x) {
        int i = idx >> 7, c = idx & 127;
        int row = row0 + i, b = row >> 9, t = row & 511;
        float val;
        if (c < 64) {
            int q = (t == 0) ? qseqs[b * 511] : sqseqs[b * 511 + t - 1];
            val = qtab[(size_t)q * 64 + c];
        } else {
            int d = c - 64; float s = 0.f;
#pragma unroll
            for (int kk = 0; kk < 4; ++kk) {
                int cc = (t == 0) ? cseqs[(size_t)(b * 511) * 4 + kk]
                                  : scseqs[(size_t)(b * 511 + t - 1) * 4 + kk];
                s += ctab[(size_t)(cc + 1) * 64 + d];
            }
            val = s * 0.25f;
        }
        xh[idx] = (_Float16)val;
    }
    __syncthreads();
    // e_emb = [x*r | x*(1-r)]
    for (int idx = tid; idx < 16 * 256; idx += blockDim.x) {
        int i = idx >> 8, c = idx & 255;
        float r = rr_s[i];
        float xv = (float)xh[i * 128 + (c & 127)];
        eh[idx] = (_Float16)((c < 128) ? xv * r : xv * (1.f - r));
    }
    __syncthreads();
    // k = x @ Wk + bk
    {
        v8f acc = {};
        wmma_tile(xh, 128, WkT, 128, 128, wid, lane, acc);
        int n = wid * 16 + (lane & 15), hi = lane >> 4;
        float bb = bk[n];
#pragma unroll
        for (int r = 0; r < 8; ++r) {
            int m = hi * 8 + r;
            float y = acc[r] + bb;
            kf[(size_t)(row0 + m) * 64 + n] = y;
            khg[(size_t)(row0 + m) * 64 + n] = (_Float16)y;
            khl[m * 64 + n] = (_Float16)y;
        }
    }
    __syncthreads();
    // v = e_emb @ Wv + bv (K=256)
    {
        v8f acc = {};
        wmma_tile(eh, 256, WvT, 256, 256, wid, lane, acc);
        int n = wid * 16 + (lane & 15), hi = lane >> 4;
        float bb = bv[n];
#pragma unroll
        for (int r = 0; r < 8; ++r) vhl[(hi * 8 + r) * 64 + n] = (_Float16)(acc[r] + bb);
    }
    __syncthreads();
    // e = sigmoid(v@We+be)
    {
        v8f acc = {};
        wmma_tile(vhl, 64, WeT, 64, 64, wid, lane, acc);
        int n = wid * 16 + (lane & 15), hi = lane >> 4;
        float bb = be[n];
#pragma unroll
        for (int r = 0; r < 8; ++r)
            ef[(size_t)(row0 + hi * 8 + r) * 64 + n] = sigm(acc[r] + bb);
    }
    // a = tanh(v@Wa+ba)
    {
        v8f acc = {};
        wmma_tile(vhl, 64, WaT, 64, 64, wid, lane, acc);
        int n = wid * 16 + (lane & 15), hi = lane >> 4;
        float bb = ba[n];
#pragma unroll
        for (int r = 0; r < 8; ++r)
            af[(size_t)(row0 + hi * 8 + r) * 64 + n] = tanhf(acc[r] + bb);
    }
    // scores = k @ Mk^T  (N padded to 64)
    {
        v8f acc = {};
        wmma_tile(khl, 64, MkT, 64, 64, wid, lane, acc);
        int n = wid * 16 + (lane & 15), hi = lane >> 4;
#pragma unroll
        for (int r = 0; r < 8; ++r) sc[(hi * 8 + r) * 64 + n] = acc[r];
    }
    __syncthreads();
    // row softmax over first 50 cols -> w
    if (tid < 16) {
        float mx = -1e30f;
        for (int m = 0; m < 50; ++m) mx = fmaxf(mx, sc[tid * 64 + m]);
        float s = 0.f;
        for (int m = 0; m < 50; ++m) { float e2 = expf(sc[tid * 64 + m] - mx); sc[tid * 64 + m] = e2; s += e2; }
        float inv = 1.f / s;
        for (int m = 0; m < 50; ++m) wf[(size_t)(row0 + tid) * 50 + m] = sc[tid * 64 + m] * inv;
    }
}

// ---------- 3. sequential memory scan ----------
// One block per batch, 64 threads; thread d owns Mv[:, d] in 50 VGPRs.
// Next-step w (LDS double buffer) and e/a (registers) are prefetched so the
// per-step global-load latency overlaps the 50-FMA register chain.
__global__ void dkvmn_scan(const float* __restrict__ Mv0, const float* __restrict__ wf,
                           const float* __restrict__ ef, const float* __restrict__ af,
                           float* __restrict__ readf, _Float16* __restrict__ readh) {
    __shared__ float wl[2][64];
    const int d = threadIdx.x;      // column 0..63
    const int b = blockIdx.x;
    float mv[50];
#pragma unroll
    for (int m = 0; m < 50; ++m) mv[m] = Mv0[m * 64 + d];

    const size_t base = (size_t)b * 512;
    if (d < 50) wl[0][d] = wf[base * 50 + d];
    float et = ef[base * 64 + d];
    float at = af[base * 64 + d];
    __syncthreads();

    for (int t = 0; t < 512; ++t) {
        const size_t row = base + t;
        const int cur = t & 1;
        float etn = 0.f, atn = 0.f;
        if (t + 1 < 512) {                       // stage t+1 while computing t
            if (d < 50) wl[cur ^ 1][d] = wf[(row + 1) * 50 + d];
            etn = ef[(row + 1) * 64 + d];
            atn = af[(row + 1) * 64 + d];
        }
        float rd = 0.f;
#pragma unroll
        for (int m = 0; m < 50; ++m) {
            float wm = wl[cur][m];               // LDS broadcast
            rd += wm * mv[m];                    // read of pre-update state
            mv[m] = mv[m] * (1.f - wm * et) + wm * at;
        }
        readf[row * 64 + d] = rd;
        readh[row * 64 + d] = (_Float16)rd;
        et = etn; at = atn;
        __syncthreads();                         // wl[cur^1] ready / wl[cur] free
    }
}

// ---------- 4. back end: f = tanh([read,k]@Wf+bf), heads, loss pieces ----------
__global__ void dkvmn_back(const _Float16* __restrict__ readh, const _Float16* __restrict__ khg,
                           const _Float16* __restrict__ WfT, const float* __restrict__ bf,
                           const float* __restrict__ kf, const float* __restrict__ readf,
                           const float* __restrict__ Wp, const float* __restrict__ bp,
                           const float* __restrict__ Wq, const float* __restrict__ bq,
                           const float* __restrict__ Ws, const float* __restrict__ bs,
                           const int* __restrict__ srseqs, const int* __restrict__ sqseqs,
                           const float* __restrict__ constp,
                           float* __restrict__ out, float* __restrict__ partials) {
    __shared__ _Float16 cat[16 * 128];
    __shared__ float ff[16 * 64];
    __shared__ float red[48];
    const int tid = threadIdx.x, lane = tid & 31, wid = tid >> 5;
    const int row0 = blockIdx.x * 16;
    const int Nm = 64 * 511;

    prefetch_bytes(WfT, 64 * 128 * 2, tid, blockDim.x);

    for (int idx = tid; idx < 16 * 128; idx += blockDim.x) {
        int i = idx >> 7, c = idx & 127;
        size_t row = (size_t)(row0 + i);
        cat[idx] = (c < 64) ? readh[row * 64 + c] : khg[row * 64 + (c - 64)];
    }
    __syncthreads();
    {
        v8f acc = {};
        wmma_tile(cat, 128, WfT, 128, 128, wid, lane, acc);
        int n = wid * 16 + (lane & 15), hi = lane >> 4;
        float bb = bf[n];
#pragma unroll
        for (int r = 0; r < 8; ++r) ff[(hi * 8 + r) * 64 + n] = tanhf(acc[r] + bb);
    }
    __syncthreads();

    if (tid < 16) {
        int row = row0 + tid, b = row >> 9, t = row & 511;
        float l0 = bp[0], l1 = bp[1], q0 = bq[0], q1 = bq[1], s0 = bs[0], s1 = bs[1];
        for (int d2 = 0; d2 < 64; ++d2) {
            float fd = ff[tid * 64 + d2];
            float kd = kf[(size_t)row * 64 + d2];
            float rd = readf[(size_t)row * 64 + d2];
            l0 += fd * Wp[d2 * 2 + 0]; l1 += fd * Wp[d2 * 2 + 1];
            q0 += kd * Wq[d2 * 2 + 0]; q1 += kd * Wq[d2 * 2 + 1];
            s0 += rd * Ws[d2 * 2 + 0]; s1 += rd * Ws[d2 * 2 + 1];
        }
        float ce_q = 0.f, ce_z = 0.f, kl = 0.f;
        if (t >= 1) {
            const float cst = constp[0];
            int j = b * 511 + (t - 1);
            int label = srseqs[j];
            int qid = sqseqs[j];
            // z_qks / z_q (=z_nde)
            float a0 = l0 + q0 + s0, a1 = l1 + q1 + s1;
            float z0 = logf(sigm(a0) + EPSV), z1 = logf(sigm(a1) + EPSV);
            float zn0 = logf(sigm(q0 + 2.f * cst) + EPSV), zn1 = logf(sigm(q1 + 2.f * cst) + EPSV);
            // CE(q_logit)
            float qm = fmaxf(q0, q1);
            float lse_q = qm + logf(expf(q0 - qm) + expf(q1 - qm));
            ce_q = lse_q - ((label == 0) ? q0 : q1);
            // CE(z_qks)
            float zm = fmaxf(z0, z1);
            float lse_z = zm + logf(expf(z0 - zm) + expf(z1 - zm));
            ce_z = lse_z - ((label == 0) ? z0 : z1);
            // KL: -softmax(z_qks) . log_softmax(z_nde)
            float p0 = expf(z0 - lse_z), p1 = expf(z1 - lse_z);
            float nm = fmaxf(zn0, zn1);
            float lse_n = nm + logf(expf(zn0 - nm) + expf(zn1 - nm));
            kl = -(p0 * (zn0 - lse_n) + p1 * (zn1 - lse_n));
            // pred = softmax(core)[1], core = z_qks - z_nde
            float c0 = z0 - zn0, c1 = z1 - zn1;
            float cm = fmaxf(c0, c1);
            float lse_c = cm + logf(expf(c0 - cm) + expf(c1 - cm));
            float pred = expf(c1 - lse_c);
            out[1 + j] = pred;
            out[1 + Nm + j] = (float)label;
            out[1 + 2 * Nm + j] = (float)(qid + 10000 * label);
        }
        red[tid * 3 + 0] = ce_q;
        red[tid * 3 + 1] = ce_z;
        red[tid * 3 + 2] = kl;
    }
    __syncthreads();
    if (tid == 0) {
        float s0 = 0.f, s1 = 0.f, s2 = 0.f;
        for (int i = 0; i < 16; ++i) { s0 += red[i * 3]; s1 += red[i * 3 + 1]; s2 += red[i * 3 + 2]; }
        partials[(size_t)blockIdx.x * 3 + 0] = s0;
        partials[(size_t)blockIdx.x * 3 + 1] = s1;
        partials[(size_t)blockIdx.x * 3 + 2] = s2;
    }
}

// ---------- 5. deterministic tree reduce -> loss ----------
__global__ void dkvmn_finalize(const float* __restrict__ partials, int nvals, float* __restrict__ out) {
    __shared__ float acc[256];
    float s = 0.f;
    for (int i = threadIdx.x; i < nvals; i += blockDim.x) s += partials[i];
    acc[threadIdx.x] = s;
    __syncthreads();
    for (int st = 128; st > 0; st >>= 1) {
        if ((int)threadIdx.x < st) acc[threadIdx.x] += acc[threadIdx.x + st];
        __syncthreads();
    }
    if (threadIdx.x == 0) out[0] = acc[0] / (float)(64 * 511);
}

extern "C" void kernel_launch(void* const* d_in, const int* in_sizes, int n_in,
                              void* d_out, int out_size, void* d_ws, size_t ws_size,
                              hipStream_t stream) {
    (void)in_sizes; (void)n_in; (void)out_size; (void)ws_size;
    const int*   qseqs  = (const int*)d_in[0];
    const int*   cseqs  = (const int*)d_in[1];
    const int*   rseqs  = (const int*)d_in[2];
    const int*   sqseqs = (const int*)d_in[3];
    const int*   scseqs = (const int*)d_in[4];
    const int*   srseqs = (const int*)d_in[5];
    // d_in[6] = smasks (all ones, unused)
    const float* Wk  = (const float*)d_in[7];
    const float* bk  = (const float*)d_in[8];
    const float* Wv  = (const float*)d_in[9];
    const float* bv  = (const float*)d_in[10];
    const float* Mk  = (const float*)d_in[11];
    const float* Mv0 = (const float*)d_in[12];
    const float* We  = (const float*)d_in[13];
    const float* be  = (const float*)d_in[14];
    const float* Wa  = (const float*)d_in[15];
    const float* ba  = (const float*)d_in[16];
    const float* Wf  = (const float*)d_in[17];
    const float* bf  = (const float*)d_in[18];
    const float* Wp  = (const float*)d_in[19];
    const float* bp  = (const float*)d_in[20];
    const float* Wq  = (const float*)d_in[21];
    const float* bq  = (const float*)d_in[22];
    const float* Ws  = (const float*)d_in[23];
    const float* bs  = (const float*)d_in[24];
    const float* ctab  = (const float*)d_in[25];
    const float* qtab  = (const float*)d_in[26];
    const float* constp = (const float*)d_in[27];

    const size_t R = 64 * 512;            // total rows
    char* ws = (char*)d_ws;
    size_t off = 0;
    auto alloc = [&](size_t bytes) -> void* {
        void* p = ws + off;
        off = (off + bytes + 255) & ~(size_t)255;
        return p;
    };
    _Float16* WkT = (_Float16*)alloc(64 * 128 * 2);
    _Float16* WvT = (_Float16*)alloc(64 * 256 * 2);
    _Float16* WeT = (_Float16*)alloc(64 * 64 * 2);
    _Float16* WaT = (_Float16*)alloc(64 * 64 * 2);
    _Float16* WfT = (_Float16*)alloc(64 * 128 * 2);
    _Float16* MkT = (_Float16*)alloc(64 * 64 * 2);
    float*    kf    = (float*)alloc(R * 64 * 4);
    _Float16* khg   = (_Float16*)alloc(R * 64 * 2);
    float*    ef    = (float*)alloc(R * 64 * 4);
    float*    af    = (float*)alloc(R * 64 * 4);
    float*    wf    = (float*)alloc(R * 50 * 4);
    float*    readf = (float*)alloc(R * 64 * 4);
    _Float16* readh = (_Float16*)alloc(R * 64 * 2);
    float*    partials = (float*)alloc(2048 * 3 * 4);

    float* out = (float*)d_out;

    dkvmn_prep<<<64, 256, 0, stream>>>(Wk, Wv, We, Wa, Wf, Mk, WkT, WvT, WeT, WaT, WfT, MkT);
    dkvmn_front<<<2048, 128, 0, stream>>>(qseqs, cseqs, rseqs, sqseqs, scseqs, srseqs,
                                          qtab, ctab, WkT, WvT, WeT, WaT, MkT,
                                          bk, bv, be, ba, kf, khg, ef, af, wf);
    dkvmn_scan<<<64, 64, 0, stream>>>(Mv0, wf, ef, af, readf, readh);
    dkvmn_back<<<2048, 128, 0, stream>>>(readh, khg, WfT, bf, kf, readf,
                                         Wp, bp, Wq, bq, Ws, bs,
                                         srseqs, sqseqs, constp, out, partials);
    dkvmn_finalize<<<1, 256, 0, stream>>>(partials, 2048 * 3, out);
}
